// CrossSliceAttention2D_2877628088964
// MI455X (gfx1250) — matrix operand order repaired and checked by reading, not compiled, not run on hardware
//
#include <hip/hip_runtime.h>
#include <hip/hip_bf16.h>

// ---------------- types ----------------
typedef __attribute__((ext_vector_type(16))) __bf16 v16bf;
typedef __attribute__((ext_vector_type(8)))  __bf16 v8bf;
typedef __attribute__((ext_vector_type(8)))  float  v8f;

#define HEADS 8
#define HD 48
#define DPAD 64
#define INNER 384
#define CQ 256
#define NTOK 2304         // 48*48
#define NBATCH 4
#define ATT_SCALE 0.14433756729740643f  // 48^-0.5

__device__ inline v16bf cat8(v8bf lo, v8bf hi) {
    return __builtin_shufflevector(lo, hi, 0,1,2,3,4,5,6,7,8,9,10,11,12,13,14,15);
}

__device__ inline v8f wmma_bf16(v16bf a, v16bf b, v8f c) {
    return __builtin_amdgcn_wmma_f32_16x16x32_bf16(false, a, false, b, (short)0, c, false, false);
}

__device__ inline float rowmax16(float v) {
    v = fmaxf(v, __shfl_xor(v, 1, 32));
    v = fmaxf(v, __shfl_xor(v, 2, 32));
    v = fmaxf(v, __shfl_xor(v, 4, 32));
    v = fmaxf(v, __shfl_xor(v, 8, 32));
    return v;
}
__device__ inline float rowsum16(float v) {
    v += __shfl_xor(v, 1, 32);
    v += __shfl_xor(v, 2, 32);
    v += __shfl_xor(v, 4, 32);
    v += __shfl_xor(v, 8, 32);
    return v;
}

// ---------------- kernel 0: zero workspace (pad lanes must be 0) -------------
__global__ void zero_ws_kernel(uint4* __restrict__ p, size_t n16) {
    size_t i = (size_t)blockIdx.x * blockDim.x + threadIdx.x;
    size_t stride = (size_t)gridDim.x * blockDim.x;
    uint4 z; z.x = 0; z.y = 0; z.z = 0; z.w = 0;
    for (; i < n16; i += stride) p[i] = z;
}

// ---------------- kernel 1: GroupNorm statistics -----------------------------
// one block per (b, group); 8 channels x 2304 pixels reduced to mean/rstd
__global__ __launch_bounds__(256)
void gn_stats_kernel(const float* __restrict__ X, float* __restrict__ stats) {
    __shared__ float ssum[256];
    __shared__ float ssq[256];
    const int bg = blockIdx.x;            // 0..127
    const int b = bg >> 5, g = bg & 31;
    const int t = threadIdx.x;
    const float* base = X + ((size_t)b * CQ + g * 8) * NTOK;
    float s = 0.f, q = 0.f;
    #pragma unroll
    for (int cc = 0; cc < 8; ++cc) {
        for (int n = t; n < NTOK; n += 256) {
            float v = base[cc * NTOK + n];
            s += v; q += v * v;
        }
    }
    ssum[t] = s; ssq[t] = q;
    __syncthreads();
    for (int off = 128; off > 0; off >>= 1) {
        if (t < off) { ssum[t] += ssum[t + off]; ssq[t] += ssq[t + off]; }
        __syncthreads();
    }
    if (t == 0) {
        const float inv = 1.f / (8.f * (float)NTOK);
        float mu = ssum[0] * inv;
        float var = ssq[0] * inv - mu * mu;
        stats[bg] = mu;
        stats[128 + bg] = rsqrtf(var + 1e-5f);
    }
}

// ---------------- kernel 2: QKV projection (WMMA GEMM) -----------------------
// out[o,n] = sum_c W[o,c] * GN?(X[c,n]) + bias[o]; written bf16 head-major.
// grid: (36 n-tiles of 64, 3 m-tiles of 128, 4 batches); 256 thr = 8 waves.
__global__ __launch_bounds__(256)
void qkv_proj_kernel(const float* __restrict__ X, const float* __restrict__ W,
                     const float* __restrict__ bias,
                     const float* __restrict__ gnw, const float* __restrict__ gnb,
                     const float* __restrict__ stats, int do_gn,
                     __bf16* __restrict__ outBuf) {
    __shared__ __align__(16) __bf16 Ws[128][32];   // [m_local][k]
    __shared__ __align__(16) __bf16 Xs[64][40];    // transposed [n_local][k], padded
    const int n0 = blockIdx.x * 64;
    const int m0 = blockIdx.y * 128;
    const int b  = blockIdx.z;
    const int t = threadIdx.x;
    const int wave = t >> 5, lane = t & 31;
    const int half = lane >> 4, l16 = lane & 15;
    v8f acc[4] = {};

    for (int k0 = 0; k0 < CQ; k0 += 32) {
        { // stage weights 128x32 -> bf16
            int r = t >> 1, jseg = (t & 1) * 16;
            const float* wp = W + (size_t)(m0 + r) * CQ + k0 + jseg;
            float4 f0 = ((const float4*)wp)[0];
            float4 f1 = ((const float4*)wp)[1];
            float4 f2 = ((const float4*)wp)[2];
            float4 f3 = ((const float4*)wp)[3];
            v8bf w0, w1;
            w0[0]=(__bf16)f0.x; w0[1]=(__bf16)f0.y; w0[2]=(__bf16)f0.z; w0[3]=(__bf16)f0.w;
            w0[4]=(__bf16)f1.x; w0[5]=(__bf16)f1.y; w0[6]=(__bf16)f1.z; w0[7]=(__bf16)f1.w;
            w1[0]=(__bf16)f2.x; w1[1]=(__bf16)f2.y; w1[2]=(__bf16)f2.z; w1[3]=(__bf16)f2.w;
            w1[4]=(__bf16)f3.x; w1[5]=(__bf16)f3.y; w1[6]=(__bf16)f3.z; w1[7]=(__bf16)f3.w;
            *(v8bf*)&Ws[r][jseg]     = w0;
            *(v8bf*)&Ws[r][jseg + 8] = w1;
        }
        { // stage activations 32x64, GN fused, transposed into LDS
            int j = t >> 3, i0 = (t & 7) * 8;
            int c = k0 + j;
            const float* xp = X + ((size_t)b * CQ + c) * NTOK + n0 + i0;
            float4 x0 = ((const float4*)xp)[0];
            float4 x1 = ((const float4*)xp)[1];
            float vals[8] = {x0.x,x0.y,x0.z,x0.w,x1.x,x1.y,x1.z,x1.w};
            if (do_gn) {
                int sg = b * 32 + (c >> 3);
                float mu = stats[sg], rs = stats[128 + sg];
                float sc = rs * gnw[c];
                float sh = gnb[c] - mu * sc;
                #pragma unroll
                for (int ii = 0; ii < 8; ++ii) vals[ii] = vals[ii] * sc + sh;
            }
            #pragma unroll
            for (int ii = 0; ii < 8; ++ii) Xs[i0 + ii][j] = (__bf16)vals[ii];
        }
        __syncthreads();
        // A fragment: rows wave*16 + (lane%16), K = (i/8)*16 + half*8 + i%8
        const __bf16* arow = &Ws[wave * 16 + l16][0];
        v16bf a = cat8(*(const v8bf*)(arow + half * 8),
                       *(const v8bf*)(arow + 16 + half * 8));
        #pragma unroll
        for (int sub = 0; sub < 4; ++sub) {
            const __bf16* brow = &Xs[sub * 16 + l16][0];
            v16bf bf = cat8(*(const v8bf*)(brow + half * 8),
                            *(const v8bf*)(brow + 16 + half * 8));
            acc[sub] = wmma_bf16(a, bf, acc[sub]);
        }
        __syncthreads();
    }
    // epilogue: bias, convert to bf16, scatter to [b,h,n,DPAD]
    #pragma unroll
    for (int sub = 0; sub < 4; ++sub) {
        int n = n0 + sub * 16 + l16;
        #pragma unroll
        for (int r = 0; r < 8; ++r) {
            int o = m0 + wave * 16 + r + half * 8;   // inner index 0..383
            float v = acc[sub][r] + bias[o];
            int h = o / HD, d = o % HD;
            outBuf[(((size_t)b * HEADS + h) * NTOK + n) * DPAD + d] = (__bf16)v;
        }
    }
}

// ---------------- kernel 3: flash attention ----------------------------------
// grid (18 q-tiles of 128, 8 heads, 4 batches); 8 waves, one 16-row slice each.
// K tile staged with CDNA5 async global->LDS DMA (ASYNCcnt path).
__global__ __launch_bounds__(256)
void attn_kernel(const __bf16* __restrict__ Qb, const __bf16* __restrict__ Kb,
                 const __bf16* __restrict__ Vb, __bf16* __restrict__ Ob) {
    __shared__ __align__(16) __bf16 Ks[64][64];     // [key][d]
    __shared__ __align__(16) __bf16 Vt[64][72];     // transposed [d][key], padded
    __shared__ __align__(16) __bf16 Ps[8][16][64];  // per-wave P staging
    const int qt = blockIdx.x, h = blockIdx.y, b = blockIdx.z;
    const int t = threadIdx.x, wave = t >> 5, lane = t & 31;
    const int half = lane >> 4, l16 = lane & 15;
    const size_t headBase = ((size_t)(b * HEADS + h)) * NTOK * DPAD;

    // Q fragments held in registers for the whole key loop (16 x 64, 2 k-steps)
    const __bf16* qrow = Qb + headBase + (size_t)(qt * 128 + wave * 16 + l16) * DPAD;
    v16bf aq[2];
    #pragma unroll
    for (int s = 0; s < 2; ++s)
        aq[s] = cat8(*(const v8bf*)(qrow + s * 32 + half * 8),
                     *(const v8bf*)(qrow + s * 32 + 16 + half * 8));

    v8f oacc[4] = {};
    float mrow[8], lrow[8], alpha[8];
    #pragma unroll
    for (int r = 0; r < 8; ++r) { mrow[r] = -1e30f; lrow[r] = 0.f; }

    const int key_s = t >> 2, seg_s = (t & 3) * 16;          // staging roles
    const unsigned ldsK = (unsigned)(uintptr_t)(&Ks[key_s][seg_s]);

    for (int kt = 0; kt < NTOK / 64; ++kt) {
        { // stage K row-major via async DMA; V transposed via VGPRs
            const __bf16* kp = Kb + headBase + (size_t)(kt * 64 + key_s) * DPAD + seg_s;
            const __bf16* vp = Vb + headBase + (size_t)(kt * 64 + key_s) * DPAD + seg_s;
            // 32B per lane: INST_OFFSET applies to both LDS and global addresses
            asm volatile("global_load_async_to_lds_b128 %0, %1, off\n\t"
                         "global_load_async_to_lds_b128 %0, %1, off offset:16"
                         :: "v"(ldsK), "v"(kp) : "memory");
            v8bf v0 = *(const v8bf*)vp, v1 = *(const v8bf*)(vp + 8);
            #pragma unroll
            for (int ii = 0; ii < 8; ++ii) {
                Vt[seg_s + ii][key_s]     = v0[ii];
                Vt[seg_s + 8 + ii][key_s] = v1[ii];
            }
            if (kt + 1 < NTOK / 64) { // hint next V tile into cache
                __builtin_prefetch(vp + 64 * DPAD, 0, 1);
            }
            asm volatile("s_wait_asynccnt 0x0" ::: "memory");
        }
        __syncthreads();

        // S = Q K^T (16 x 64 per wave): contraction over padded d = 64
        v8f s[4] = {};
        #pragma unroll
        for (int sub = 0; sub < 4; ++sub) {
            const __bf16* krow = &Ks[sub * 16 + l16][0];
            v16bf b0 = cat8(*(const v8bf*)(krow + half * 8),
                            *(const v8bf*)(krow + 16 + half * 8));
            v16bf b1 = cat8(*(const v8bf*)(krow + 32 + half * 8),
                            *(const v8bf*)(krow + 32 + 16 + half * 8));
            s[sub] = wmma_bf16(aq[0], b0, s[sub]);
            s[sub] = wmma_bf16(aq[1], b1, s[sub]);
        }

        // online softmax (each row spread over the 16 lanes of one half -> xor 1/2/4/8)
        #pragma unroll
        for (int sub = 0; sub < 4; ++sub)
            #pragma unroll
            for (int r = 0; r < 8; ++r) s[sub][r] *= ATT_SCALE;
        #pragma unroll
        for (int r = 0; r < 8; ++r) {
            float mx = fmaxf(fmaxf(s[0][r], s[1][r]), fmaxf(s[2][r], s[3][r]));
            mx = rowmax16(mx);
            float mnew = fmaxf(mrow[r], mx);
            float al = __expf(mrow[r] - mnew);
            mrow[r] = mnew;
            float rs = 0.f;
            #pragma unroll
            for (int sub = 0; sub < 4; ++sub) {
                float p = __expf(s[sub][r] - mnew);
                s[sub][r] = p;
                rs += p;
            }
            rs = rowsum16(rs);
            lrow[r] = al * lrow[r] + rs;
            alpha[r] = al;
        }
        #pragma unroll
        for (int dsub = 0; dsub < 4; ++dsub)
            #pragma unroll
            for (int r = 0; r < 8; ++r) oacc[dsub][r] *= alpha[r];

        // re-fragment P: C-layout -> LDS -> A-layout (same-wave DS is in-order)
        #pragma unroll
        for (int sub = 0; sub < 4; ++sub)
            #pragma unroll
            for (int r = 0; r < 8; ++r)
                Ps[wave][r + half * 8][sub * 16 + l16] = (__bf16)s[sub][r];

        const __bf16* prow = &Ps[wave][l16][0];
        v16bf ap[2];
        #pragma unroll
        for (int ss = 0; ss < 2; ++ss)
            ap[ss] = cat8(*(const v8bf*)(prow + ss * 32 + half * 8),
                          *(const v8bf*)(prow + ss * 32 + 16 + half * 8));

        // O += P V (contraction over 64 keys, 2 k-steps)
        #pragma unroll
        for (int dsub = 0; dsub < 4; ++dsub) {
            const __bf16* vrow = &Vt[dsub * 16 + l16][0];
            v16bf b0 = cat8(*(const v8bf*)(vrow + half * 8),
                            *(const v8bf*)(vrow + 16 + half * 8));
            v16bf b1 = cat8(*(const v8bf*)(vrow + 32 + half * 8),
                            *(const v8bf*)(vrow + 32 + 16 + half * 8));
            oacc[dsub] = wmma_bf16(ap[0], b0, oacc[dsub]);
            oacc[dsub] = wmma_bf16(ap[1], b1, oacc[dsub]);
        }
        __syncthreads();
    }

    #pragma unroll
    for (int r = 0; r < 8; ++r) lrow[r] = 1.f / lrow[r];
    #pragma unroll
    for (int dsub = 0; dsub < 4; ++dsub) {
        int d = dsub * 16 + l16;
        #pragma unroll
        for (int r = 0; r < 8; ++r) {
            int n = qt * 128 + wave * 16 + r + half * 8;
            Ob[headBase + (size_t)n * DPAD + d] = (__bf16)(oacc[dsub][r] * lrow[r]);
        }
    }
}

// ---------------- kernel 4: output projection + residual ---------------------
// out[c,n] = sum_o Wo[c,o]*O[o,n] + bo[c] + q_feat[c,n]
__global__ __launch_bounds__(256)
void out_proj_kernel(const __bf16* __restrict__ Ob, const float* __restrict__ Wo,
                     const float* __restrict__ bo, const float* __restrict__ resid,
                     float* __restrict__ out) {
    __shared__ __align__(16) __bf16 Ws[128][32];
    __shared__ __align__(16) __bf16 Xs[64][40];   // transposed [n_local][k]
    const int n0 = blockIdx.x * 64;
    const int m0 = blockIdx.y * 128;
    const int b  = blockIdx.z;
    const int t = threadIdx.x;
    const int wave = t >> 5, lane = t & 31;
    const int half = lane >> 4, l16 = lane & 15;
    v8f acc[4] = {};

    for (int k0 = 0; k0 < INNER; k0 += 32) {
        { // Wo tile [256,384]
            int r = t >> 1, jseg = (t & 1) * 16;
            const float* wp = Wo + (size_t)(m0 + r) * INNER + k0 + jseg;
            float4 f0 = ((const float4*)wp)[0];
            float4 f1 = ((const float4*)wp)[1];
            float4 f2 = ((const float4*)wp)[2];
            float4 f3 = ((const float4*)wp)[3];
            v8bf w0, w1;
            w0[0]=(__bf16)f0.x; w0[1]=(__bf16)f0.y; w0[2]=(__bf16)f0.z; w0[3]=(__bf16)f0.w;
            w0[4]=(__bf16)f1.x; w0[5]=(__bf16)f1.y; w0[6]=(__bf16)f1.z; w0[7]=(__bf16)f1.w;
            w1[0]=(__bf16)f2.x; w1[1]=(__bf16)f2.y; w1[2]=(__bf16)f2.z; w1[3]=(__bf16)f2.w;
            w1[4]=(__bf16)f3.x; w1[5]=(__bf16)f3.y; w1[6]=(__bf16)f3.z; w1[7]=(__bf16)f3.w;
            *(v8bf*)&Ws[r][jseg]     = w0;
            *(v8bf*)&Ws[r][jseg + 8] = w1;
        }
        { // O tile: gather inner index o = k0+j -> (h,d), stride-64 over n
            int j = t >> 3, nl0 = (t & 7) * 8;
            int o = k0 + j;
            int h = o / HD, d = o % HD;
            const __bf16* op = Ob + (((size_t)(b * HEADS + h)) * NTOK + n0 + nl0) * DPAD + d;
            #pragma unroll
            for (int ii = 0; ii < 8; ++ii) Xs[nl0 + ii][j] = op[ii * DPAD];
        }
        __syncthreads();
        const __bf16* arow = &Ws[wave * 16 + l16][0];
        v16bf a = cat8(*(const v8bf*)(arow + half * 8),
                       *(const v8bf*)(arow + 16 + half * 8));
        #pragma unroll
        for (int sub = 0; sub < 4; ++sub) {
            const __bf16* brow = &Xs[sub * 16 + l16][0];
            v16bf bf = cat8(*(const v8bf*)(brow + half * 8),
                            *(const v8bf*)(brow + 16 + half * 8));
            acc[sub] = wmma_bf16(a, bf, acc[sub]);
        }
        __syncthreads();
    }
    #pragma unroll
    for (int sub = 0; sub < 4; ++sub) {
        int n = n0 + sub * 16 + l16;
        #pragma unroll
        for (int r = 0; r < 8; ++r) {
            int c = m0 + wave * 16 + r + half * 8;
            size_t idx = ((size_t)(b * CQ + c)) * NTOK + n;
            out[idx] = acc[sub][r] + bo[c] + resid[idx];
        }
    }
}

// ---------------- host entry -------------------------------------------------
extern "C" void kernel_launch(void* const* d_in, const int* in_sizes, int n_in,
                              void* d_out, int out_size, void* d_ws, size_t ws_size,
                              hipStream_t stream) {
    const float* q_feat  = (const float*)d_in[0];
    const float* kv_feat = (const float*)d_in[1];
    const float* gn_w    = (const float*)d_in[2];
    const float* gn_b    = (const float*)d_in[3];
    const float* wq      = (const float*)d_in[4];
    const float* bq      = (const float*)d_in[5];
    const float* wk      = (const float*)d_in[6];
    const float* bk      = (const float*)d_in[7];
    const float* wv      = (const float*)d_in[8];
    const float* bv      = (const float*)d_in[9];
    const float* wo      = (const float*)d_in[10];
    const float* bo      = (const float*)d_in[11];
    float* out = (float*)d_out;

    char* ws = (char*)d_ws;
    float* stats = (float*)ws;                                   // 256 floats
    const size_t HEADBUF = (size_t)NBATCH * HEADS * NTOK * DPAD * 2; // bytes (bf16)
    __bf16* Qb   = (__bf16*)(ws + 4096);
    __bf16* Kb   = (__bf16*)(ws + 4096 + HEADBUF);
    __bf16* Vb   = (__bf16*)(ws + 4096 + 2 * HEADBUF);
    __bf16* Obuf = (__bf16*)(ws + 4096 + 3 * HEADBUF);

    zero_ws_kernel<<<2048, 256, 0, stream>>>((uint4*)(ws + 4096), (4 * HEADBUF) / 16);
    gn_stats_kernel<<<128, 256, 0, stream>>>(q_feat, stats);
    qkv_proj_kernel<<<dim3(36, 3, 4), 256, 0, stream>>>(q_feat,  wq, bq, gn_w, gn_b, stats, 1, Qb);
    qkv_proj_kernel<<<dim3(36, 3, 4), 256, 0, stream>>>(kv_feat, wk, bk, gn_w, gn_b, stats, 0, Kb);
    qkv_proj_kernel<<<dim3(36, 3, 4), 256, 0, stream>>>(kv_feat, wv, bv, gn_w, gn_b, stats, 0, Vb);
    attn_kernel<<<dim3(18, 8, 4), 256, 0, stream>>>(Qb, Kb, Vb, Obuf);
    out_proj_kernel<<<dim3(36, 2, 4), 256, 0, stream>>>(Obuf, wo, bo, q_feat, out);
}
